// StackedRNN_85048942396147
// MI455X (gfx1250) — compile-verified, matrix operand order checked by reading, not run on hardware
//
#include <hip/hip_runtime.h>
#include <hip/hip_bf16.h>
#include <stdint.h>

// Problem dimensions (match reference)
#define B_    64
#define T_    256
#define IN_   512
#define H_    1024
#define G4H   4096            // 4*H

// Launch geometry for the persistent kernel
#define NBLK   64
#define NTHR   256
#define NWAVES (NBLK * NTHR / 32)   // 512 wave32 waves
#define NTILES_N 256                 // 4096/16
#define KT_IH0  (IN_ / 32)           // 16 k-steps of 32
#define KT_H    (H_ / 32)            // 32 k-steps of 32

typedef __attribute__((ext_vector_type(16))) __bf16 v16bf;
typedef __attribute__((ext_vector_type(8)))  __bf16 v8bf;
typedef __attribute__((ext_vector_type(8)))  float  v8f;

// ---------------- workspace layout (bytes) ----------------
static constexpr size_t AL(size_t x) { return (x + 255) & ~size_t(255); }
static constexpr size_t BAR_OFF   = 0;                                       // 2 x u32 barrier
static constexpr size_t BSUM0_OFF = AL(BAR_OFF + 2 * sizeof(unsigned));      // combined bias L0
static constexpr size_t BSUM1_OFF = AL(BSUM0_OFF + (size_t)G4H * 4);         // combined bias L1
static constexpr size_t PWIH0_OFF = AL(BSUM1_OFF + (size_t)G4H * 4);         // packed W_ih0 bf16
static constexpr size_t PWHH0_OFF = AL(PWIH0_OFF + (size_t)G4H * IN_ * 2);
static constexpr size_t PWIH1_OFF = AL(PWHH0_OFF + (size_t)G4H * H_ * 2);
static constexpr size_t PWHH1_OFF = AL(PWIH1_OFF + (size_t)G4H * H_ * 2);
static constexpr size_t XSTG_OFF  = AL(PWHH1_OFF + (size_t)G4H * H_ * 2);    // x_t staged bf16
static constexpr size_t H0BF_OFF  = AL(XSTG_OFF + (size_t)B_ * IN_ * 2);
static constexpr size_t H1BF_OFF  = AL(H0BF_OFF + (size_t)B_ * H_ * 2);
static constexpr size_t C0_OFF    = AL(H1BF_OFF + (size_t)B_ * H_ * 2);
static constexpr size_t C1_OFF    = AL(C0_OFF + (size_t)B_ * H_ * 4);
static constexpr size_t H0F_OFF   = AL(C1_OFF + (size_t)B_ * H_ * 4);
static constexpr size_t H1F_OFF   = AL(H0F_OFF + (size_t)B_ * H_ * 4);
static constexpr size_t G0_OFF    = AL(H1F_OFF + (size_t)B_ * H_ * 4);       // gates L0 f32
static constexpr size_t G1_OFF    = AL(G0_OFF + (size_t)B_ * G4H * 4);       // gates L1 f32

// ---------------- helpers ----------------
__device__ __forceinline__ float fast_sigmoid(float x) {
    return 1.0f / (1.0f + __expf(-x));
}
__device__ __forceinline__ float fast_tanh(float x) {
    float cx = fminf(15.0f, fmaxf(-15.0f, x));
    float e  = __expf(2.0f * cx);
    return (e - 1.0f) / (e + 1.0f);
}

// A-fragment (16x32 bf16, row-major activations, stride ldA):
// lane<16 : M=lane,    elems 0..7 -> K 0..7,  elems 8..15 -> K 16..23
// lane>=16: M=lane-16, elems 0..7 -> K 8..15, elems 8..15 -> K 24..31
__device__ __forceinline__ v16bf ld_afrag(const __bf16* p) {
    union { v16bf v; v8bf h[2]; } u;
    u.h[0] = *(const v8bf*)(p);
    u.h[1] = *(const v8bf*)(p + 16);
    return u.v;
}
// B-fragment: pre-packed contiguous 16 bf16 per lane
__device__ __forceinline__ v16bf ld_bfrag(const __bf16* p) {
    union { v16bf v; v8bf h[2]; } u;
    u.h[0] = *(const v8bf*)(p);
    u.h[1] = *(const v8bf*)(p + 8);
    return u.v;
}

// Dual-M-tile GEMM: one B-fragment load feeds two WMMAs (tiles mt0 and mt0+1,
// same nt). Software-pipelined one k-step ahead so fragment loads overlap the
// matrix ops instead of forcing s_wait_loadcnt 0 before every WMMA.
__device__ __forceinline__ void gemm2(v8f& acc0, v8f& acc1,
                                      const __bf16* __restrict__ act, int ldA,
                                      const __bf16* __restrict__ packW, int nKt,
                                      int mt0, int nt, int lane) {
    const __bf16* ab0 = act + (size_t)(mt0 * 16 + (lane & 15)) * ldA + ((lane & 16) ? 8 : 0);
    const __bf16* ab1 = ab0 + (size_t)16 * ldA;
    const __bf16* bb  = packW + ((size_t)nt * nKt) * 512 + ((size_t)lane << 4);

    v16bf bc  = ld_bfrag(bb);
    v16bf a0c = ld_afrag(ab0);
    v16bf a1c = ld_afrag(ab1);
#pragma unroll
    for (int kt = 0; kt < nKt; ++kt) {
        v16bf bn, a0n, a1n;
        if (kt + 1 < nKt) {
            bn  = ld_bfrag(bb + (size_t)(kt + 1) * 512);
            a0n = ld_afrag(ab0 + (kt + 1) * 32);
            a1n = ld_afrag(ab1 + (kt + 1) * 32);
        }
        acc0 = __builtin_amdgcn_wmma_f32_16x16x32_bf16(false, a0c, false, bc,
                                                       (short)0, acc0, false, false);
        acc1 = __builtin_amdgcn_wmma_f32_16x16x32_bf16(false, a1c, false, bc,
                                                       (short)0, acc1, false, false);
        if (kt + 1 < nKt) { bc = bn; a0c = a0n; a1c = a1n; }
    }
}

// D layout: lane L -> col = L%16, VGPR r -> row = r + (L>=16 ? 8 : 0)
__device__ __forceinline__ void store_tile(float* __restrict__ g, int mt, int nt,
                                           int lane, v8f acc) {
    int col  = nt * 16 + (lane & 15);
    int row0 = mt * 16 + ((lane & 16) ? 8 : 0);
    float* p = g + (size_t)row0 * G4H + col;
#pragma unroll
    for (int r = 0; r < 8; ++r) p[(size_t)r * G4H] = acc[r];
}

__device__ __forceinline__ void stage_x(const float* __restrict__ x, int t,
                                        __bf16* __restrict__ xstg, int gtid) {
    for (int i = gtid; i < B_ * IN_; i += NBLK * NTHR) {
        int b = i >> 9, k = i & (IN_ - 1);
        xstg[i] = (__bf16)x[((size_t)b * T_ + t) * IN_ + k];
    }
}

// grid-wide barrier: monotonic arrival counter + generation flag in global memory
__device__ __forceinline__ void gbar(unsigned* cnt, unsigned* gen, unsigned& phase) {
    __syncthreads();
    unsigned ph = ++phase;
    if (threadIdx.x == 0) {
        __threadfence();
        unsigned old = __hip_atomic_fetch_add(cnt, 1u, __ATOMIC_ACQ_REL,
                                              __HIP_MEMORY_SCOPE_AGENT);
        if (old == ph * NBLK - 1u) {
            __hip_atomic_store(gen, ph, __ATOMIC_RELEASE, __HIP_MEMORY_SCOPE_AGENT);
        } else {
            while (__hip_atomic_load(gen, __ATOMIC_ACQUIRE,
                                     __HIP_MEMORY_SCOPE_AGENT) < ph) {
                __builtin_amdgcn_s_sleep(2);
            }
        }
        __threadfence();
    }
    __syncthreads();
}

// ---------------- kernels ----------------
__global__ void lstm_init(uint8_t* __restrict__ ws) {
    int gid = blockIdx.x * blockDim.x + threadIdx.x;
    int n   = gridDim.x * blockDim.x;
    if (gid == 0) {
        unsigned* b = (unsigned*)(ws + BAR_OFF);
        b[0] = 0u; b[1] = 0u;
    }
    float*  c0  = (float*)(ws + C0_OFF);
    float*  c1  = (float*)(ws + C1_OFF);
    float*  h0f = (float*)(ws + H0F_OFF);
    float*  h1f = (float*)(ws + H1F_OFF);
    __bf16* h0b = (__bf16*)(ws + H0BF_OFF);
    __bf16* h1b = (__bf16*)(ws + H1BF_OFF);
    for (int i = gid; i < B_ * H_; i += n) {
        c0[i] = 0.f; c1[i] = 0.f; h0f[i] = 0.f; h1f[i] = 0.f;
        h0b[i] = (__bf16)0.f; h1b[i] = (__bf16)0.f;
    }
}

// Pack W (4096 x K, row-major f32) into per-wave WMMA B-fragments (bf16).
// Fragment f = (nt*nKt + kt)*32 + lane, 16 bf16 each:
// lane<16 : N=nt*16+lane,    K = kt*32 + {0..7, 16..23}
// lane>=16: N=nt*16+lane-16, K = kt*32 + {8..15, 24..31}
__global__ void lstm_pack(const float* __restrict__ W, int K, __bf16* __restrict__ dst) {
    int nKt   = K >> 5;
    int total = NTILES_N * nKt * 32;
    for (int f = blockIdx.x * blockDim.x + threadIdx.x; f < total;
         f += gridDim.x * blockDim.x) {
        int lane = f & 31;
        int r    = f >> 5;
        int kt   = r % nKt;
        int nt   = r / nKt;
        int n    = nt * 16 + (lane & 15);
        int kb   = kt * 32 + ((lane & 16) ? 8 : 0);
        __bf16*      d = dst + (size_t)f * 16;
        const float* s = W + (size_t)n * K + kb;
#pragma unroll
        for (int e = 0; e < 8; ++e) d[e] = (__bf16)s[e];
#pragma unroll
        for (int e = 0; e < 8; ++e) d[8 + e] = (__bf16)s[16 + e];
    }
}

__global__ void lstm_bias(const float* __restrict__ bi0, const float* __restrict__ bh0,
                          const float* __restrict__ bi1, const float* __restrict__ bh1,
                          uint8_t* __restrict__ ws) {
    int i = blockIdx.x * blockDim.x + threadIdx.x;
    if (i < G4H) {
        ((float*)(ws + BSUM0_OFF))[i] = bi0[i] + bh0[i];
        ((float*)(ws + BSUM1_OFF))[i] = bi1[i] + bh1[i];
    }
}

__global__ void __launch_bounds__(NTHR, 1)
lstm_persistent(const float* __restrict__ x, float* __restrict__ out,
                uint8_t* __restrict__ ws) {
    const int tid  = threadIdx.x;
    const int gtid = blockIdx.x * NTHR + tid;
    const int lane = tid & 31;
    const int wave = gtid >> 5;

    // tile assignment: each wave owns two M-tiles of one N-tile
    const int nt  = wave & 255;        // 0..255
    const int mt0 = (wave >> 8) * 2;   // 0 or 2 -> tiles (mt0,nt),(mt0+1,nt)

    unsigned* bar_cnt = (unsigned*)(ws + BAR_OFF);
    unsigned* bar_gen = bar_cnt + 1;
    const float*  bs0  = (const float*)(ws + BSUM0_OFF);
    const float*  bs1  = (const float*)(ws + BSUM1_OFF);
    const __bf16* pwih0 = (const __bf16*)(ws + PWIH0_OFF);
    const __bf16* pwhh0 = (const __bf16*)(ws + PWHH0_OFF);
    const __bf16* pwih1 = (const __bf16*)(ws + PWIH1_OFF);
    const __bf16* pwhh1 = (const __bf16*)(ws + PWHH1_OFF);
    __bf16* xstg = (__bf16*)(ws + XSTG_OFF);
    __bf16* h0b  = (__bf16*)(ws + H0BF_OFF);
    __bf16* h1b  = (__bf16*)(ws + H1BF_OFF);
    float*  c0   = (float*)(ws + C0_OFF);
    float*  c1   = (float*)(ws + C1_OFF);
    float*  h0f  = (float*)(ws + H0F_OFF);
    float*  h1f  = (float*)(ws + H1F_OFF);
    float*  g0   = (float*)(ws + G0_OFF);
    float*  g1   = (float*)(ws + G1_OFF);

    unsigned phase = 0;

    // prologue: stage x_0 as bf16
    stage_x(x, 0, xstg, gtid);
    gbar(bar_cnt, bar_gen, phase);

    for (int t = 0; t < T_; ++t) {
        // ---- GEMM layer 0: gates0 = x_t @ W_ih0^T + h0 @ W_hh0^T ----
        {
            v8f acc0 = {}, acc1 = {};
            gemm2(acc0, acc1, xstg, IN_, pwih0, KT_IH0, mt0, nt, lane);
            gemm2(acc0, acc1, h0b,  H_,  pwhh0, KT_H,   mt0, nt, lane);
            store_tile(g0, mt0,     nt, lane, acc0);
            store_tile(g0, mt0 + 1, nt, lane, acc1);
        }
        gbar(bar_cnt, bar_gen, phase);

        // ---- elementwise layer 0 ----
        for (int i = gtid; i < B_ * H_; i += NBLK * NTHR) {
            int b = i >> 10, u = i & (H_ - 1);
            const float* gr = g0 + (size_t)b * G4H;
            float gi = gr[u]            + bs0[u];
            float gf = gr[H_ + u]       + bs0[H_ + u];
            float gg = gr[2 * H_ + u]   + bs0[2 * H_ + u];
            float go = gr[3 * H_ + u]   + bs0[3 * H_ + u];
            float I = fast_sigmoid(gi);
            float F = fast_sigmoid(gf);
            float G = fast_tanh(gg);
            float O = fast_sigmoid(go);
            float c = F * c0[i] + I * G;
            float h = O * fast_tanh(c);
            c0[i] = c; h0f[i] = h; h0b[i] = (__bf16)h;
        }
        gbar(bar_cnt, bar_gen, phase);

        // ---- GEMM layer 1: gates1 = h0_new @ W_ih1^T + h1 @ W_hh1^T ----
        {
            v8f acc0 = {}, acc1 = {};
            gemm2(acc0, acc1, h0b, H_, pwih1, KT_H, mt0, nt, lane);
            gemm2(acc0, acc1, h1b, H_, pwhh1, KT_H, mt0, nt, lane);
            store_tile(g1, mt0,     nt, lane, acc0);
            store_tile(g1, mt0 + 1, nt, lane, acc1);
        }
        // stage next step's x while gates1 settle
        if (t + 1 < T_) stage_x(x, t + 1, xstg, gtid);
        gbar(bar_cnt, bar_gen, phase);

        // ---- elementwise layer 1 + output write ----
        for (int i = gtid; i < B_ * H_; i += NBLK * NTHR) {
            int b = i >> 10, u = i & (H_ - 1);
            const float* gr = g1 + (size_t)b * G4H;
            float gi = gr[u]            + bs1[u];
            float gf = gr[H_ + u]       + bs1[H_ + u];
            float gg = gr[2 * H_ + u]   + bs1[2 * H_ + u];
            float go = gr[3 * H_ + u]   + bs1[3 * H_ + u];
            float I = fast_sigmoid(gi);
            float F = fast_sigmoid(gf);
            float G = fast_tanh(gg);
            float O = fast_sigmoid(go);
            float c = F * c1[i] + I * G;
            float h = O * fast_tanh(c);
            c1[i] = c; h1f[i] = h; h1b[i] = (__bf16)h;
            out[((size_t)b * T_ + t) * H_ + u] = h;
        }
        gbar(bar_cnt, bar_gen, phase);
    }

    // epilogue: final hidden states (h0, c0, h1, c1) after the outputs
    float* tail = out + (size_t)B_ * T_ * H_;
    for (int i = gtid; i < B_ * H_; i += NBLK * NTHR) {
        tail[i]               = h0f[i];
        tail[B_ * H_ + i]     = c0[i];
        tail[2 * B_ * H_ + i] = h1f[i];
        tail[3 * B_ * H_ + i] = c1[i];
    }
}

// ---------------- host entry ----------------
extern "C" void kernel_launch(void* const* d_in, const int* in_sizes, int n_in,
                              void* d_out, int out_size, void* d_ws, size_t ws_size,
                              hipStream_t stream) {
    (void)in_sizes; (void)n_in; (void)out_size; (void)ws_size;
    const float* x    = (const float*)d_in[0];
    const float* Wih0 = (const float*)d_in[1];
    const float* Whh0 = (const float*)d_in[2];
    const float* bih0 = (const float*)d_in[3];
    const float* bhh0 = (const float*)d_in[4];
    const float* Wih1 = (const float*)d_in[5];
    const float* Whh1 = (const float*)d_in[6];
    const float* bih1 = (const float*)d_in[7];
    const float* bhh1 = (const float*)d_in[8];
    uint8_t* ws  = (uint8_t*)d_ws;
    float*   out = (float*)d_out;

    lstm_init<<<256, 256, 0, stream>>>(ws);
    lstm_pack<<<512, 256, 0, stream>>>(Wih0, IN_, (__bf16*)(ws + PWIH0_OFF));
    lstm_pack<<<1024, 256, 0, stream>>>(Whh0, H_, (__bf16*)(ws + PWHH0_OFF));
    lstm_pack<<<1024, 256, 0, stream>>>(Wih1, H_, (__bf16*)(ws + PWIH1_OFF));
    lstm_pack<<<1024, 256, 0, stream>>>(Whh1, H_, (__bf16*)(ws + PWHH1_OFF));
    lstm_bias<<<16, 256, 0, stream>>>(bih0, bhh0, bih1, bhh1, ws);
    lstm_persistent<<<NBLK, NTHR, 0, stream>>>(x, out, ws);
}
